// MultiHeadAttention_50251117363504
// MI455X (gfx1250) — matrix operand order repaired and checked by reading, not compiled, not run on hardware
//
#include <hip/hip_runtime.h>
#include <hip/hip_bf16.h>
#include <math.h>

// ---------------------------------------------------------------------------
// MultiHeadAttention forward for MI455X (gfx1250), wave32 + WMMA bf16 + TDM.
//   x:[2,4096,768] fp32; W*:[768,768]; b*:[768]; out:[2,4096,768] fp32
// Pipeline: cvt(x,W*)->bf16 | GEMM QKV | flash-attn per (b,h) | GEMM out-proj
// Tiles are streamed global->LDS with tensor_load_to_lds (TENSORcnt), double
// buffered so the DMA overlaps the WMMA stream.
// ---------------------------------------------------------------------------

#define EMB    768
#define HEADS  8
#define HD     96
#define SEQ    4096
#define BATCH  2
#define NROWS  (BATCH * SEQ)           // 8192
#define SCALE_F 0.10206207261596577f   // 96^-0.5

typedef __attribute__((ext_vector_type(16))) __bf16 v16bf;
typedef __attribute__((ext_vector_type(8)))  __bf16 bf8;
typedef __attribute__((ext_vector_type(8)))  float  v8f;
typedef __attribute__((ext_vector_type(4)))  unsigned int u32x4;
typedef __attribute__((ext_vector_type(4)))  int i32x4;
typedef __attribute__((ext_vector_type(8)))  int i32x8;

#if __has_builtin(__builtin_amdgcn_tensor_load_to_lds)
#define HAVE_TDM 1
#else
#define HAVE_TDM 0   // host pass / toolchains without TDM builtin: manual staging
#endif

__device__ __forceinline__ v16bf cat16(bf8 lo, bf8 hi) {
  return __builtin_shufflevector(lo, hi, 0,1,2,3,4,5,6,7,8,9,10,11,12,13,14,15);
}

__device__ __forceinline__ v8f wmma_bf16(v16bf a, v16bf b, v8f c) {
  return __builtin_amdgcn_wmma_f32_16x16x32_bf16(
      false, a, false, b, (short)0, c, false, false);
}

__device__ __forceinline__ unsigned ldsOffset(const void* p) {
  // LDS aperture: low 32 bits of the flat address are the LDS byte offset.
  return (unsigned)(unsigned long long)p;
}

#if HAVE_TDM
// 2D tile DMA: global (row stride strideElems, bf16) -> LDS contiguous rows,
// optional LDS padding every (2<<padInt) DWORDs of (padAmt+1) DWORDs.
__device__ __forceinline__ void tdm_load_2d(unsigned lds, unsigned long long gaddr,
                                            unsigned tile0, unsigned tile1,
                                            unsigned long long strideElems,
                                            unsigned padInt, unsigned padAmt,
                                            bool pad) {
  u32x4 g0;
  g0[0] = 1u;                                           // count=1, user mode
  g0[1] = lds;                                          // lds_addr (bytes)
  g0[2] = (unsigned)(gaddr & 0xFFFFFFFFull);            // global_addr[31:0]
  g0[3] = (unsigned)((gaddr >> 32) & 0x01FFFFFFull) | (2u << 30);  // [56:32]|type=2
  i32x8 g1;
  g1[0] = (int)((1u << 16) |                            // data_size=1 -> 2 bytes
                (pad ? ((1u << 20) | (padInt << 22) | (padAmt << 25)) : 0u));
  g1[1] = (int)((tile0 & 0xFFFFu) << 16);               // tensor_dim0[15:0]@48
  g1[2] = (int)(((tile0 >> 16) & 0xFFFFu) | ((tile1 & 0xFFFFu) << 16));
  g1[3] = (int)(((tile1 >> 16) & 0xFFFFu) | ((tile0 & 0xFFFFu) << 16)); // tile_dim0@112
  g1[4] = (int)(tile1 & 0xFFFFu);                       // tile_dim1; tile_dim2=0
  g1[5] = (int)(strideElems & 0xFFFFFFFFull);           // tensor_dim0_stride[31:0]
  g1[6] = (int)((strideElems >> 32) & 0xFFFFull);
  g1[7] = 0;
  i32x4 z4 = {};
#if defined(__clang_major__) && __clang_major__ >= 23
  i32x8 z8 = {};
  __builtin_amdgcn_tensor_load_to_lds(g0, g1, z4, z4, z8, 0);
#else
  __builtin_amdgcn_tensor_load_to_lds(g0, g1, z4, z4, 0);
#endif
}
#endif  // HAVE_TDM

// ------------------------------ fp32 -> bf16 --------------------------------
__global__ void cvt_f32_bf16(const float* __restrict__ in,
                             __bf16* __restrict__ out, int n) {
  int i = blockIdx.x * blockDim.x + threadIdx.x;
  int stride = gridDim.x * blockDim.x;
  for (; i < n; i += stride) out[i] = (__bf16)in[i];
}

// ------------------------- GEMM: out = A * W^T + b --------------------------
// Block = 8 waves, tile 256x64; each wave: 32 rows x 64 cols (8 accumulators).
// K stepped by 32; A/B tiles TDM double-buffered through LDS.
template <bool OUT_BF16>
__global__ __launch_bounds__(256)
void gemm_bias(const __bf16* __restrict__ A, const __bf16* __restrict__ W,
               const float* __restrict__ bias, void* __restrict__ outv) {
  __shared__ __bf16 As[2][256][40];   // 32 used + 8 pad (TDM pad: 16DW/4DW)
  __shared__ __bf16 Bs[2][64][48];    // 32 used + 16 pad (TDM pad: 16DW/8DW)

  const int tid  = threadIdx.x;
  const int wv   = tid >> 5;
  const int lane = tid & 31;
  const int hh   = lane >> 4;
  const int l16  = lane & 15;

  const int n0 = blockIdx.x * 64;
  const int m0 = blockIdx.y * 256;

  v8f acc[2][4] = {};

#if HAVE_TDM
  const unsigned la0 = ldsOffset(&As[0][0][0]), la1 = ldsOffset(&As[1][0][0]);
  const unsigned lb0 = ldsOffset(&Bs[0][0][0]), lb1 = ldsOffset(&Bs[1][0][0]);
  if (wv == 0) {
    tdm_load_2d(la0, (unsigned long long)(A + (size_t)m0 * EMB), 32, 256, EMB, 3, 3, true);
    tdm_load_2d(lb0, (unsigned long long)(W + (size_t)n0 * EMB), 32, 64,  EMB, 3, 7, true);
  }
#endif

  int buf = 0;
  for (int k0 = 0; k0 < EMB; k0 += 32, buf ^= 1) {
    __syncthreads();   // previous compute done -> buf^1 free for DMA
#if HAVE_TDM
    if (wv == 0) {
      if (k0 + 32 < EMB) {
        tdm_load_2d(buf ? la0 : la1,
                    (unsigned long long)(A + (size_t)m0 * EMB + k0 + 32), 32, 256, EMB, 3, 3, true);
        tdm_load_2d(buf ? lb0 : lb1,
                    (unsigned long long)(W + (size_t)n0 * EMB + k0 + 32), 32, 64,  EMB, 3, 7, true);
        __builtin_amdgcn_s_wait_tensorcnt(2);  // older pair (current buf) done
      } else {
        __builtin_amdgcn_s_wait_tensorcnt(0);
      }
    }
#else
    for (int c = tid; c < 1024; c += 256) {        // A tile 256x32
      int r = c >> 2, ko = (c & 3) * 8;
      *(bf8*)&As[buf][r][ko] = *(const bf8*)&A[(size_t)(m0 + r) * EMB + k0 + ko];
    }
    { int r = tid >> 2, ko = (tid & 3) * 8;        // B tile 64x32
      *(bf8*)&Bs[buf][r][ko] = *(const bf8*)&W[(size_t)(n0 + r) * EMB + k0 + ko]; }
#endif
    __syncthreads();

    #pragma unroll
    for (int q = 0; q < 2; ++q) {
      int arow = wv * 32 + q * 16 + l16;
      bf8 alo = *(const bf8*)&As[buf][arow][16 * hh];
      bf8 ahi = *(const bf8*)&As[buf][arow][16 * hh + 8];
      v16bf a = cat16(alo, ahi);
      #pragma unroll
      for (int t = 0; t < 4; ++t) {
        v16bf b = *(const v16bf*)&Bs[buf][t * 16 + l16][16 * hh];
        acc[q][t] = wmma_bf16(a, b, acc[q][t]);
      }
    }
  }

  #pragma unroll
  for (int q = 0; q < 2; ++q)
    #pragma unroll
    for (int t = 0; t < 4; ++t) {
      int col = n0 + t * 16 + l16;
      float bv = bias[col];
      #pragma unroll
      for (int r = 0; r < 8; ++r) {
        int row = m0 + wv * 32 + q * 16 + r + 8 * hh;
        float v = acc[q][t][r] + bv;
        if (OUT_BF16) ((__bf16*)outv)[(size_t)row * EMB + col] = (__bf16)v;
        else          ((float*)outv)[(size_t)row * EMB + col] = v;
      }
    }
}

// ------------------------------ flash attention -----------------------------
#define BQ 128
#define BK 64

__global__ __launch_bounds__(256)
void flash_attn(const __bf16* __restrict__ Q, const __bf16* __restrict__ K,
                const __bf16* __restrict__ V, __bf16* __restrict__ O) {
  __shared__ __bf16 Qs[BQ][HD];        // 24 KB (TDM)
  __shared__ __bf16 Ks[2][BK][HD];     // 24 KB (TDM, double buffered)
  __shared__ __bf16 Vt[HD][BK];        // 12 KB (manual: needs transpose)
  __shared__ __bf16 Ps[8][16][BK];     // 16 KB (per-wave P bounce)

  const int tid  = threadIdx.x;
  const int wv   = tid >> 5;
  const int lane = tid & 31;
  const int hh   = lane >> 4;
  const int l16  = lane & 15;

  const int q0 = blockIdx.x * BQ;
  const int bh = blockIdx.y;
  const int b  = bh >> 3, h = bh & 7;
  const size_t base = (size_t)b * SEQ * EMB + h * HD;

#if HAVE_TDM
  const unsigned lq  = ldsOffset(&Qs[0][0]);
  const unsigned lk0 = ldsOffset(&Ks[0][0][0]), lk1 = ldsOffset(&Ks[1][0][0]);
  if (wv == 0) {
    tdm_load_2d(lq,  (unsigned long long)(Q + base + (size_t)q0 * EMB), HD, BQ, EMB, 0, 0, false);
    tdm_load_2d(lk0, (unsigned long long)(K + base),                    HD, BK, EMB, 0, 0, false);
  }
#else
  for (int c = tid; c < BQ * 12; c += 256) {
    int r = c / 12, ko = (c % 12) * 8;
    *(bf8*)&Qs[r][ko] = *(const bf8*)&Q[base + (size_t)(q0 + r) * EMB + ko];
  }
#endif

  float mrow[8], lrow[8];
  v8f o[6] = {};
  #pragma unroll
  for (int r = 0; r < 8; ++r) { mrow[r] = -3.0e38f; lrow[r] = 0.0f; }

  int kb = 0;
  for (int kt = 0; kt < SEQ / BK; ++kt, kb ^= 1) {
    const int k0r = kt * BK;
    __syncthreads();   // prev compute done -> Vt and Ks[kb^1] free

    // stage V tile transposed: Vt[d][j] (VGPR path; TDM cannot transpose)
    for (int c = tid; c < BK * 12; c += 256) {
      int r = c / 12, ko = (c % 12) * 8;
      bf8 v = *(const bf8*)&V[base + (size_t)(k0r + r) * EMB + ko];
      #pragma unroll
      for (int e = 0; e < 8; ++e) Vt[ko + e][r] = v[e];
    }
#if HAVE_TDM
    if (wv == 0) {
      if (k0r + BK < SEQ) {
        tdm_load_2d(kb ? lk0 : lk1,
                    (unsigned long long)(K + base + (size_t)(k0r + BK) * EMB), HD, BK, EMB, 0, 0, false);
        __builtin_amdgcn_s_wait_tensorcnt(1);   // current K tile (older) done
      } else {
        __builtin_amdgcn_s_wait_tensorcnt(0);
      }
    }
#else
    for (int c = tid; c < BK * 12; c += 256) {
      int r = c / 12, ko = (c % 12) * 8;
      *(bf8*)&Ks[kb][r][ko] = *(const bf8*)&K[base + (size_t)(k0r + r) * EMB + ko];
    }
#endif
    if (k0r + BK < SEQ)
      __builtin_prefetch(&V[base + (size_t)(k0r + BK + (tid >> 2)) * EMB], 0, 0);
    __syncthreads();

    // ---- S = Q * K^T (wave's 16 rows x 64 keys) ----
    v8f s[4] = {};
    #pragma unroll
    for (int kc = 0; kc < 3; ++kc) {
      int kk = kc * 32;
      bf8 alo = *(const bf8*)&Qs[wv * 16 + l16][kk + 16 * hh];
      bf8 ahi = *(const bf8*)&Qs[wv * 16 + l16][kk + 16 * hh + 8];
      v16bf a = cat16(alo, ahi);
      #pragma unroll
      for (int t = 0; t < 4; ++t) {
        v16bf bm = *(const v16bf*)&Ks[kb][t * 16 + l16][kk + 16 * hh];
        s[t] = wmma_bf16(a, bm, s[t]);
      }
    }
    #pragma unroll
    for (int t = 0; t < 4; ++t) s[t] = s[t] * SCALE_F;

    // ---- online softmax; acc element r -> row (r + 8*hh), col l16 ----
    float tm[8], alpha[8];
    #pragma unroll
    for (int r = 0; r < 8; ++r) {
      float m = fmaxf(fmaxf(s[0][r], s[1][r]), fmaxf(s[2][r], s[3][r]));
      #pragma unroll
      for (int msk = 1; msk < 16; msk <<= 1) m = fmaxf(m, __shfl_xor(m, msk, 32));
      tm[r] = m;
    }
    #pragma unroll
    for (int r = 0; r < 8; ++r) {
      float mn = fmaxf(mrow[r], tm[r]);
      alpha[r] = __expf(mrow[r] - mn);
      mrow[r] = mn;
    }
    #pragma unroll
    for (int t = 0; t < 4; ++t)
      #pragma unroll
      for (int r = 0; r < 8; ++r) s[t][r] = __expf(s[t][r] - mrow[r]);
    #pragma unroll
    for (int r = 0; r < 8; ++r) {
      float v = s[0][r] + s[1][r] + s[2][r] + s[3][r];
      #pragma unroll
      for (int msk = 1; msk < 16; msk <<= 1) v += __shfl_xor(v, msk, 32);
      lrow[r] = lrow[r] * alpha[r] + v;
    }
    #pragma unroll
    for (int t = 0; t < 6; ++t)
      #pragma unroll
      for (int r = 0; r < 8; ++r) o[t][r] *= alpha[r];

    // ---- bounce P through LDS into A-fragment layout (same wave only) ----
    #pragma unroll
    for (int t = 0; t < 4; ++t)
      #pragma unroll
      for (int r = 0; r < 8; ++r)
        Ps[wv][r + 8 * hh][t * 16 + l16] = (__bf16)s[t][r];

    // ---- O += P * V ----
    #pragma unroll
    for (int kc = 0; kc < 2; ++kc) {
      int kk = kc * 32;
      bf8 alo = *(const bf8*)&Ps[wv][l16][kk + 16 * hh];
      bf8 ahi = *(const bf8*)&Ps[wv][l16][kk + 16 * hh + 8];
      v16bf a = cat16(alo, ahi);
      #pragma unroll
      for (int t = 0; t < 6; ++t) {
        v16bf bm = *(const v16bf*)&Vt[t * 16 + l16][kk + 16 * hh];
        o[t] = wmma_bf16(a, bm, o[t]);
      }
    }
  }

  #pragma unroll
  for (int t = 0; t < 6; ++t)
    #pragma unroll
    for (int r = 0; r < 8; ++r) {
      int row = q0 + wv * 16 + r + 8 * hh;
      O[base + (size_t)row * EMB + t * 16 + l16] = (__bf16)(o[t][r] / lrow[r]);
    }
}

// --------------------------------- launch -----------------------------------
extern "C" void kernel_launch(void* const* d_in, const int* in_sizes, int n_in,
                              void* d_out, int out_size, void* d_ws, size_t ws_size,
                              hipStream_t stream) {
  const float* x  = (const float*)d_in[0];
  const float* Wq = (const float*)d_in[1];
  const float* bq = (const float*)d_in[2];
  const float* Wk = (const float*)d_in[3];
  const float* bk = (const float*)d_in[4];
  const float* Wv = (const float*)d_in[5];
  const float* bv = (const float*)d_in[6];
  const float* Wo = (const float*)d_in[7];
  const float* bo = (const float*)d_in[8];
  float* out = (float*)d_out;

  char* ws = (char*)d_ws;
  size_t off = 0;
  auto take = [&](size_t bytes) -> char* {
    char* p = ws + off;
    off += (bytes + 255) & ~(size_t)255;
    return p;
  };
  __bf16* xb  = (__bf16*)take((size_t)NROWS * EMB * 2);
  __bf16* Wqb = (__bf16*)take((size_t)EMB * EMB * 2);
  __bf16* Wkb = (__bf16*)take((size_t)EMB * EMB * 2);
  __bf16* Wvb = (__bf16*)take((size_t)EMB * EMB * 2);
  __bf16* Wob = (__bf16*)take((size_t)EMB * EMB * 2);
  __bf16* Qb  = (__bf16*)take((size_t)NROWS * EMB * 2);
  __bf16* Kb  = (__bf16*)take((size_t)NROWS * EMB * 2);
  __bf16* Vb  = (__bf16*)take((size_t)NROWS * EMB * 2);
  __bf16* Ob  = (__bf16*)take((size_t)NROWS * EMB * 2);

  cvt_f32_bf16<<<2048, 256, 0, stream>>>(x,  xb,  NROWS * EMB);
  cvt_f32_bf16<<<512, 256, 0, stream>>>(Wq, Wqb, EMB * EMB);
  cvt_f32_bf16<<<512, 256, 0, stream>>>(Wk, Wkb, EMB * EMB);
  cvt_f32_bf16<<<512, 256, 0, stream>>>(Wv, Wvb, EMB * EMB);
  cvt_f32_bf16<<<512, 256, 0, stream>>>(Wo, Wob, EMB * EMB);

  dim3 gg(EMB / 64, NROWS / 256);
  gemm_bias<true><<<gg, 256, 0, stream>>>(xb, Wqb, bq, Qb);
  gemm_bias<true><<<gg, 256, 0, stream>>>(xb, Wkb, bk, Kb);
  gemm_bias<true><<<gg, 256, 0, stream>>>(xb, Wvb, bv, Vb);

  dim3 ga(SEQ / BQ, BATCH * HEADS);
  flash_attn<<<ga, 256, 0, stream>>>(Qb, Kb, Vb, Ob);

  gemm_bias<false><<<gg, 256, 0, stream>>>(Ob, Wob, bo, out);
}
